// PreT_Attention_77034533421518
// MI455X (gfx1250) — matrix operand order repaired and checked by reading, not compiled, not run on hardware
//
#include <hip/hip_runtime.h>
#include <cstddef>

// ---------------------------------------------------------------------------
// PreT_Attention for MI455X (gfx1250, wave32, WMMA bf16).
// B=16, N=1024, C=1024, H=16, D=64, P=16, M=P+N=1040 padded to 1056.
// ---------------------------------------------------------------------------

typedef __bf16 bf16;
typedef __attribute__((ext_vector_type(16))) __bf16 v16bf;
typedef __attribute__((ext_vector_type(8)))  __bf16 v8bf;
typedef __attribute__((ext_vector_type(8)))  float  v8f;

#define Bn   16
#define Nn   1024
#define Cn   1024
#define Hn   16
#define Dn   64
#define Pn   16
#define Mreal 1040
#define Mpad  1056
#define SCALE 0.125f

// ---- WMMA helpers ----------------------------------------------------------

__device__ __forceinline__ v8f wmma_bf(v16bf a, v16bf b, v8f c) {
  // D(16x16,f32) = A(16x32,bf16) x B(32x16,bf16) + C
  return __builtin_amdgcn_wmma_f32_16x16x32_bf16(false, a, false, b,
                                                 (short)0, c, false, false);
}

// A-matrix 16x32 bf16 fragment. Memory layout: [row][k], row-major, `stride`
// elements per row. Per ISA: lane row = lane&15, half = lane>>4,
// elements 0..7 -> k = half*8 .. half*8+7 ; elements 8..15 -> +16.
__device__ __forceinline__ v16bf frag_a(const bf16* p, int lane, int stride) {
  const int row = lane & 15, half = lane >> 4;
  const bf16* q = p + row * stride + half * 8;
  v16bf a;
  ((v8bf*)&a)[0] = *(const v8bf*)(q);
  ((v8bf*)&a)[1] = *(const v8bf*)(q + 16);
  return a;
}

// B-matrix 32x16 bf16 fragment (K=32, N=16). Memory layout: [n][k] (k
// contiguous per column), `stride` elements per column. Lane col = lane&15,
// half = lane>>4, elements 0..7 -> k = half*8 .. half*8+7 ; elements 8..15
// -> +16 (same k order as frag_a).
__device__ __forceinline__ v16bf frag_b(const bf16* p, int lane, int stride) {
  const int col = lane & 15, half = lane >> 4;
  const bf16* q = p + col * stride + half * 8;
  v16bf b;
  ((v8bf*)&b)[0] = *(const v8bf*)(q);
  ((v8bf*)&b)[1] = *(const v8bf*)(q + 16);
  return b;
}

// ---------------------------------------------------------------------------
// Shared GEMM tile compute: 128x128 block tile, 8 waves as 4(rows)x2(cols).
// Each wave: 32 rows x 64 cols = 2 A-frags x 4 B-frags -> 8 WMMAs per k-step.
// All fragments preloaded before the WMMA chain (one dscnt wait, 8 WMMAs).
// ---------------------------------------------------------------------------
__device__ __forceinline__ void tile_mma(const bf16* As, const bf16* Bs,
                                         int wr, int wc, int lane,
                                         v8f acc[2][4]) {
  v16bf a0 = frag_a(As + (wr * 32 + 0) * 32, lane, 32);
  v16bf a1 = frag_a(As + (wr * 32 + 16) * 32, lane, 32);
  v16bf b0 = frag_b(Bs + (wc * 64 + 0) * 32, lane, 32);
  v16bf b1 = frag_b(Bs + (wc * 64 + 16) * 32, lane, 32);
  v16bf b2 = frag_b(Bs + (wc * 64 + 32) * 32, lane, 32);
  v16bf b3 = frag_b(Bs + (wc * 64 + 48) * 32, lane, 32);
  acc[0][0] = wmma_bf(a0, b0, acc[0][0]);
  acc[0][1] = wmma_bf(a0, b1, acc[0][1]);
  acc[0][2] = wmma_bf(a0, b2, acc[0][2]);
  acc[0][3] = wmma_bf(a0, b3, acc[0][3]);
  acc[1][0] = wmma_bf(a1, b0, acc[1][0]);
  acc[1][1] = wmma_bf(a1, b1, acc[1][1]);
  acc[1][2] = wmma_bf(a1, b2, acc[1][2]);
  acc[1][3] = wmma_bf(a1, b3, acc[1][3]);
}

// Stage A tile: 128 rows x 32 k of f32 -> bf16 LDS [m][k].
// Thread: 16 contiguous f32 -> two 16B LDS stores.
__device__ __forceinline__ void stage_a_f32(const float* src_row, bf16* As,
                                            int tid, int src_stride) {
  const int r = tid >> 1, koff = (tid & 1) * 16;
  const float* src = src_row + (size_t)r * src_stride + koff;
  v8bf p0, p1;
#pragma unroll
  for (int i = 0; i < 8; ++i) p0[i] = (bf16)src[i];
#pragma unroll
  for (int i = 0; i < 8; ++i) p1[i] = (bf16)src[i + 8];
  *(v8bf*)&As[r * 32 + koff + 0] = p0;
  *(v8bf*)&As[r * 32 + koff + 8] = p1;
}

// Stage B tile transposed: w[k][n] (f32, row-major, `src_stride`) ->
// Bs[n][k] bf16, n in [0,128), k in [0,32). Thread owns one n and 16 k's:
// 16 coalesced global loads (consecutive n across threads) + two 16B stores.
__device__ __forceinline__ void stage_b_f32(const float* w_kn, bf16* Bs,
                                            int tid, int src_stride) {
  const int n = tid & 127, kg = (tid >> 7) * 16;
  const float* src = w_kn + (size_t)kg * src_stride + n;
  v8bf p0, p1;
#pragma unroll
  for (int i = 0; i < 8; ++i) p0[i] = (bf16)src[(size_t)i * src_stride];
#pragma unroll
  for (int i = 0; i < 8; ++i) p1[i] = (bf16)src[(size_t)(i + 8) * src_stride];
  *(v8bf*)&Bs[n * 32 + kg + 0] = p0;
  *(v8bf*)&Bs[n * 32 + kg + 8] = p1;
}

// ---------------------------------------------------------------------------
// Kernel 1: qkv = x @ w_qkv, scatter into q/k/v bf16 buffers.
// grid = (3072/128, 16384/128), block = 256 (8 waves).
// ---------------------------------------------------------------------------
__global__ __launch_bounds__(256) void qkv_gemm_kernel(
    const float* __restrict__ x, const float* __restrict__ w,
    bf16* __restrict__ qb, bf16* __restrict__ kb, bf16* __restrict__ vb) {
  __shared__ bf16 As[128 * 32];  // [m][k]
  __shared__ bf16 Bs[128 * 32];  // [n][k]
  const int tid = threadIdx.x;
  const int wave = tid >> 5, lane = tid & 31, half = lane >> 4;
  const int wr = wave >> 1, wc = wave & 1;
  const int c0 = blockIdx.x * 128;
  const int r0 = blockIdx.y * 128;

  v8f acc[2][4] = {};

  for (int kk = 0; kk < Cn; kk += 32) {
    stage_a_f32(x + (size_t)r0 * Cn + kk, As, tid, Cn);
    stage_b_f32(w + (size_t)kk * (3 * Cn) + c0, Bs, tid, 3 * Cn);
    __syncthreads();
    tile_mma(As, Bs, wr, wc, lane, acc);
    __syncthreads();
  }

  // scatter: 3C axis decomposes as (which, h, d) = (c>>10, (c>>6)&15, c&63)
#pragma unroll
  for (int sr = 0; sr < 2; ++sr) {
#pragma unroll
    for (int f = 0; f < 4; ++f) {
      const int colbase = c0 + wc * 64 + f * 16;
      const int which = colbase >> 10, h = (colbase >> 6) & 15;
      const int d = (colbase & 63) + (lane & 15);
#pragma unroll
      for (int j = 0; j < 8; ++j) {
        const int grow = r0 + wr * 32 + sr * 16 + j + 8 * half;  // b*N + n
        const int b = grow >> 10, n = grow & 1023;
        const bf16 val = (bf16)acc[sr][f][j];
        if (which == 0)
          qb[(((size_t)(b * Hn + h)) * Nn + n) * Dn + d] = val;
        else if (which == 1)
          kb[(((size_t)(b * Hn + h)) * Mpad + Pn + n) * Dn + d] = val;
        else
          vb[(((size_t)(b * Hn + h)) * Mpad + Pn + n) * Dn + d] = val;
      }
    }
  }
}

// ---------------------------------------------------------------------------
// Kernel 2: per-(batch, k/v) stats -> affine {scale, shift} for the prefix.
// grid = 32 (b*2+which), block = 256. Deterministic tree reduction.
// ---------------------------------------------------------------------------
__global__ __launch_bounds__(256) void stats_kernel(
    const bf16* __restrict__ kb, const bf16* __restrict__ vb,
    const float* __restrict__ prompt, float* __restrict__ stats) {
  const int tid = threadIdx.x;
  const int b = blockIdx.x >> 1, which = blockIdx.x & 1;
  const bf16* src = which ? vb : kb;

  float s = 0.f, ss = 0.f;
  for (int i = tid; i < Hn * Nn * Dn; i += 256) {
    const int h = i >> 16, n = (i >> 6) & 1023, d = i & 63;
    const float xv = (float)src[(((size_t)(b * Hn + h)) * Mpad + Pn + n) * Dn + d];
    s += xv; ss += xv * xv;
  }
  float ps = 0.f, pss = 0.f;
  const float* pr = prompt + (size_t)(b * 2 + which) * (Pn * Hn * Dn);
  for (int i = tid; i < Pn * Hn * Dn; i += 256) {
    const float xv = pr[i];
    ps += xv; pss += xv * xv;
  }

  __shared__ float red[4][256];
  red[0][tid] = s; red[1][tid] = ss; red[2][tid] = ps; red[3][tid] = pss;
  __syncthreads();
  for (int off = 128; off > 0; off >>= 1) {
    if (tid < off)
#pragma unroll
      for (int q = 0; q < 4; ++q) red[q][tid] += red[q][tid + off];
    __syncthreads();
  }
  if (tid == 0) {
    const float Nr = (float)(Hn * Nn * Dn), Np = (float)(Pn * Hn * Dn);
    const float rs = red[0][0], rss = red[1][0], qs = red[2][0], qss = red[3][0];
    const float r_mean = rs / Nr;
    const float r_std = sqrtf(fmaxf((rss - rs * rs / Nr) / (Nr - 1.f), 0.f)) + 1e-6f;
    const float p_mean = qs / Np;
    const float p_std = sqrtf(fmaxf((qss - qs * qs / Np) / (Np - 1.f), 0.f)) + 1e-6f;
    const float scale = r_std / p_std;
    stats[blockIdx.x * 2 + 0] = scale;
    stats[blockIdx.x * 2 + 1] = r_mean - p_mean * scale;
  }
}

// ---------------------------------------------------------------------------
// Kernel 3: write normalized prefixes into K/V heads, zero the pad rows.
// grid = 32 (b*2+which), block = 256.
// ---------------------------------------------------------------------------
__global__ __launch_bounds__(256) void prefix_kernel(
    const float* __restrict__ prompt, const float* __restrict__ stats,
    bf16* __restrict__ kb, bf16* __restrict__ vb) {
  const int tid = threadIdx.x;
  const int b = blockIdx.x >> 1, which = blockIdx.x & 1;
  bf16* dst = which ? vb : kb;
  const float scale = stats[blockIdx.x * 2 + 0];
  const float shift = stats[blockIdx.x * 2 + 1];
  const float* pr = prompt + (size_t)(b * 2 + which) * (Pn * Hn * Dn);

  for (int i = tid; i < Pn * Hn * Dn; i += 256) {  // (p,h,d)
    const int p = i >> 10, h = (i >> 6) & 15, d = i & 63;
    dst[(((size_t)(b * Hn + h)) * Mpad + p) * Dn + d] = (bf16)(pr[i] * scale + shift);
  }
  for (int i = tid; i < Hn * (Mpad - Mreal) * Dn; i += 256) {  // (h,mr,d)
    const int h = i >> 10, mr = (i >> 6) & 15, d = i & 63;
    dst[(((size_t)(b * Hn + h)) * Mpad + Mreal + mr) * Dn + d] = (bf16)0.f;
  }
}

// ---------------------------------------------------------------------------
// Kernel 4: flash attention. grid = (B*H, N/128), block = 256 (8 waves).
// Each wave owns a 16-row Q tile; 33 key-tiles of 32 (last half-masked).
// ---------------------------------------------------------------------------
__global__ __launch_bounds__(256) void flash_attn_kernel(
    const bf16* __restrict__ qb, const bf16* __restrict__ kb,
    const bf16* __restrict__ vb, bf16* __restrict__ attn_out) {
  __shared__ bf16 Ks[32 * 64];        // [key][d]
  __shared__ bf16 Vs[64 * 32];        // [d][key]  (transposed for B-frags)
  __shared__ float Ssc[8][16][32];    // per-wave score tile
  __shared__ bf16  Pb[8][16][32];     // per-wave prob tile (A-frag layout)
  __shared__ float Alp[8][16];
  __shared__ float Linv[8][16];

  const int tid = threadIdx.x;
  const int wave = tid >> 5, lane = tid & 31, half = lane >> 4;
  const int bh = blockIdx.x;                 // b*H + h
  const int b = bh >> 4, h = bh & 15;
  const int n0 = blockIdx.y * 128;

  // Load this wave's Q tile (16 rows x 64) directly into two A fragments.
  const int qrow = n0 + wave * 16 + (lane & 15);
  const bf16* qbase = qb + ((size_t)bh * Nn + qrow) * Dn;
  v16bf qa[2];
#pragma unroll
  for (int ch = 0; ch < 2; ++ch) {
    const bf16* p = qbase + ch * 32 + half * 8;
    ((v8bf*)&qa[ch])[0] = *(const v8bf*)(p);
    ((v8bf*)&qa[ch])[1] = *(const v8bf*)(p + 16);
  }

  v8f o[4] = {};
  float m_i = -3.0e38f, l_i = 0.0f;  // valid in lanes 0..15 (row = lane)
  const bf16* kbh = kb + (size_t)bh * Mpad * Dn;
  const bf16* vbh = vb + (size_t)bh * Mpad * Dn;

  for (int t = 0; t < Mpad / 32; ++t) {
    const int m_base = t * 32;
    __syncthreads();  // previous iteration done reading Ks/Vs
    {
      const int key = tid >> 3, doff = (tid & 7) * 8;
      *(uint4*)&Ks[key * 64 + doff] =
          *(const uint4*)(kbh + (size_t)(m_base + key) * Dn + doff);
      // V transposed: thread owns d=tid&63, key-group (tid>>6)*8.
      const int d = tid & 63, kg = (tid >> 6) * 8;
      const bf16* vsrc = vbh + (size_t)(m_base + kg) * Dn + d;
      v8bf pv;
#pragma unroll
      for (int i = 0; i < 8; ++i) pv[i] = vsrc[(size_t)i * Dn];
      *(v8bf*)&Vs[d * 32 + kg] = pv;
    }
    __syncthreads();

    // S = Q * K^T  (two 16x16 subtiles, K-dim 64 as two k32 chunks)
    {
      v16bf k00 = frag_b(Ks + 0 * 16 * 64 + 0, lane, 64);
      v16bf k01 = frag_b(Ks + 0 * 16 * 64 + 32, lane, 64);
      v16bf k10 = frag_b(Ks + 1 * 16 * 64 + 0, lane, 64);
      v16bf k11 = frag_b(Ks + 1 * 16 * 64 + 32, lane, 64);
      v8f s0 = {}, s1 = {};
      s0 = wmma_bf(qa[0], k00, s0);
      s0 = wmma_bf(qa[1], k01, s0);
      s1 = wmma_bf(qa[0], k10, s1);
      s1 = wmma_bf(qa[1], k11, s1);
      const int key0 = m_base + (lane & 15);
      const float madd0 = (key0 < Mreal) ? 0.0f : -3.0e38f;
      const float madd1 = (key0 + 16 < Mreal) ? 0.0f : -3.0e38f;
#pragma unroll
      for (int j = 0; j < 8; ++j) {
        Ssc[wave][j + 8 * half][lane & 15]        = s0[j] * SCALE + madd0;
        Ssc[wave][j + 8 * half][16 + (lane & 15)] = s1[j] * SCALE + madd1;
      }
    }

    // online softmax: lanes 0..15 each own one row
    if (lane < 16) {
      float rm = m_i;
#pragma unroll 4
      for (int c = 0; c < 32; ++c) rm = fmaxf(rm, Ssc[wave][lane][c]);
      const float alpha = __expf(m_i - rm);
      float ls = 0.0f;
#pragma unroll 4
      for (int c = 0; c < 32; ++c) {
        const float p = __expf(Ssc[wave][lane][c] - rm);
        ls += p;
        Pb[wave][lane][c] = (bf16)p;
      }
      l_i = l_i * alpha + ls;
      m_i = rm;
      Alp[wave][lane] = alpha;
    }

    // rescale accumulators and do O += P * V
    float aj[8];
#pragma unroll
    for (int j = 0; j < 8; ++j) aj[j] = Alp[wave][j + 8 * half];
#pragma unroll
    for (int f = 0; f < 4; ++f)
#pragma unroll
      for (int j = 0; j < 8; ++j) o[f][j] *= aj[j];

    const v16bf pa = frag_a(&Pb[wave][0][0], lane, 32);
    v16bf vb0 = frag_b(&Vs[0 * 16 * 32], lane, 32);
    v16bf vb1 = frag_b(&Vs[1 * 16 * 32], lane, 32);
    v16bf vb2 = frag_b(&Vs[2 * 16 * 32], lane, 32);
    v16bf vb3 = frag_b(&Vs[3 * 16 * 32], lane, 32);
    o[0] = wmma_bf(pa, vb0, o[0]);
    o[1] = wmma_bf(pa, vb1, o[1]);
    o[2] = wmma_bf(pa, vb2, o[2]);
    o[3] = wmma_bf(pa, vb3, o[3]);
  }

  if (lane < 16) Linv[wave][lane] = 1.0f / l_i;

  // out layout (B, N, H*D) bf16 for the projection GEMM
#pragma unroll
  for (int f = 0; f < 4; ++f) {
#pragma unroll
    for (int j = 0; j < 8; ++j) {
      const int row = n0 + wave * 16 + j + 8 * half;
      const float val = o[f][j] * Linv[wave][j + 8 * half];
      attn_out[((size_t)b * Nn + row) * Cn + h * Dn + f * 16 + (lane & 15)] =
          (bf16)val;
    }
  }
}

// ---------------------------------------------------------------------------
// Kernel 5: out = attn @ w_proj + b_proj (f32 output).
// grid = (1024/128, 16384/128), block = 256 (8 waves).
// ---------------------------------------------------------------------------
__global__ __launch_bounds__(256) void proj_gemm_kernel(
    const bf16* __restrict__ attn, const float* __restrict__ w,
    const float* __restrict__ bias, float* __restrict__ out) {
  __shared__ bf16 As[128 * 32];  // [m][k]
  __shared__ bf16 Bs[128 * 32];  // [n][k]
  const int tid = threadIdx.x;
  const int wave = tid >> 5, lane = tid & 31, half = lane >> 4;
  const int wr = wave >> 1, wc = wave & 1;
  const int c0 = blockIdx.x * 128;
  const int r0 = blockIdx.y * 128;

  v8f acc[2][4] = {};

  for (int kk = 0; kk < Cn; kk += 32) {
    {  // stage A tile (already bf16): 16 elems = two 16B copies
      const int r = tid >> 1, koff = (tid & 1) * 16;
      const bf16* src = attn + (size_t)(r0 + r) * Cn + kk + koff;
      *(uint4*)&As[r * 32 + koff + 0] = *(const uint4*)(src + 0);
      *(uint4*)&As[r * 32 + koff + 8] = *(const uint4*)(src + 8);
    }
    stage_b_f32(w + (size_t)kk * Cn + c0, Bs, tid, Cn);
    __syncthreads();
    tile_mma(As, Bs, wr, wc, lane, acc);
    __syncthreads();
  }

#pragma unroll
  for (int sr = 0; sr < 2; ++sr) {
#pragma unroll
    for (int f = 0; f < 4; ++f) {
      const int col = c0 + wc * 64 + f * 16 + (lane & 15);
      const float bv = bias[col];
#pragma unroll
      for (int j = 0; j < 8; ++j) {
        const int grow = r0 + wr * 32 + sr * 16 + j + 8 * half;
        out[(size_t)grow * Cn + col] = acc[sr][f][j] + bv;
      }
    }
  }
}

// ---------------------------------------------------------------------------
// Host launcher
// ---------------------------------------------------------------------------
extern "C" void kernel_launch(void* const* d_in, const int* in_sizes, int n_in,
                              void* d_out, int out_size, void* d_ws, size_t ws_size,
                              hipStream_t stream) {
  (void)in_sizes; (void)n_in; (void)out_size; (void)ws_size;
  const float* x      = (const float*)d_in[0];  // (B,N,C)
  const float* prompt = (const float*)d_in[1];  // (B,2,P,H,D)
  const float* w_qkv  = (const float*)d_in[2];  // (C,3C)
  const float* w_proj = (const float*)d_in[3];  // (C,C)
  const float* b_proj = (const float*)d_in[4];  // (C,)
  float* out = (float*)d_out;                   // (B,N,C)

  // Workspace layout (bf16 unless noted):
  //   qb   (B,H,N,D)     16,777,216 elems
  //   kb   (B,H,Mpad,D)  17,301,504 elems
  //   vb   (B,H,Mpad,D)  17,301,504 elems
  //   attn (B,N,C)       16,777,216 elems
  //   stats 64 f32
  bf16* qb   = (bf16*)d_ws;
  bf16* kb   = qb + (size_t)Bn * Hn * Nn * Dn;
  bf16* vb   = kb + (size_t)Bn * Hn * Mpad * Dn;
  bf16* attn = vb + (size_t)Bn * Hn * Mpad * Dn;
  float* stats = (float*)(attn + (size_t)Bn * Nn * Cn);

  qkv_gemm_kernel<<<dim3(3 * Cn / 128, Bn * Nn / 128), 256, 0, stream>>>(
      x, w_qkv, qb, kb, vb);
  stats_kernel<<<dim3(Bn * 2), 256, 0, stream>>>(kb, vb, prompt, stats);
  prefix_kernel<<<dim3(Bn * 2), 256, 0, stream>>>(prompt, stats, kb, vb);
  flash_attn_kernel<<<dim3(Bn * Hn, Nn / 128), 256, 0, stream>>>(
      qb, kb, vb, attn);
  proj_gemm_kernel<<<dim3(Cn / 128, Bn * Nn / 128), 256, 0, stream>>>(
      attn, w_proj, b_proj, out);
}
